// AlexNet_24850680775268
// MI455X (gfx1250) — 8-bit matrix operand order repaired and checked by reading, compiled, not run on hardware
//
#include <hip/hip_runtime.h>
#include <stdint.h>

typedef int   v8i __attribute__((ext_vector_type(8)));
typedef float v8f __attribute__((ext_vector_type(8)));
typedef float v2f __attribute__((ext_vector_type(2)));

#define BN_EPS 1e-5f

__device__ __forceinline__ float bn_apply(float x, const float* __restrict__ bnp, int C, int c) {
    float g = bnp[c];
    float b = bnp[C + c];
    float m = bnp[2 * C + c];
    float v = bnp[3 * C + c];
    return (x - m) * (g * rsqrtf(v + BN_EPS)) + b;
}

__global__ void k_binw(const float* __restrict__ w, signed char* __restrict__ o, int n) {
    int i = blockIdx.x * 256 + threadIdx.x;
    if (i < n) o[i] = (w[i] >= 0.0f) ? (signed char)1 : (signed char)-1;
}

__global__ void k_binw_conv(const float* __restrict__ w, signed char* __restrict__ o,
                            int Cin, int KH, int KW, int total) {
    int i = blockIdx.x * 256 + threadIdx.x;
    if (i >= total) return;
    int ci = i % Cin; int t = i / Cin;
    int kw = t % KW;  t /= KW;
    int kh = t % KH;  int co = t / KH;
    float v = w[((co * Cin + ci) * KH + kh) * KW + kw];
    o[i] = (v >= 0.0f) ? (signed char)1 : (signed char)-1;
}

__global__ void k_binw_lin1(const float* __restrict__ w, signed char* __restrict__ o, int total) {
    int i = blockIdx.x * 256 + threadIdx.x;
    if (i >= total) return;
    int col = i % 9216;
    int m = i / 9216;
    int c = col % 256;
    int hw = col / 256;
    float v = w[(long)m * 9216 + c * 36 + hw];
    o[i] = (v >= 0.0f) ? (signed char)1 : (signed char)-1;
}

__global__ void k_prep_w1(const float* __restrict__ w, float* __restrict__ o, int total) {
    int i = blockIdx.x * 256 + threadIdx.x;
    if (i >= total) return;
    int k = i % 384;
    int m = i / 384;
    o[i] = (m < 96 && k < 363) ? w[m * 363 + k] : 0.0f;
}

__global__ void k_prep_w3(const float* __restrict__ w, float* __restrict__ o, int total) {
    int i = blockIdx.x * 256 + threadIdx.x;
    if (i >= total) return;
    int k = i % 4096;
    int m = i / 4096;
    o[i] = (m < 1000) ? w[(long)m * 4096 + k] : 0.0f;
}

__global__ __launch_bounds__(256) void k_conv1_gemm(
    const float* __restrict__ x, const float* __restrict__ w1r, float* __restrict__ out) {
    __shared__ __align__(16) float Asf[64 * 36];
    __shared__ __align__(16) float Bsf[128 * 36];

    const int tid = threadIdx.x;
    const int lane = tid & 31;
    const int wid = tid >> 5;
    const int wave_m = wid & 1;
    const int wave_n = wid >> 1;
    const int n0 = blockIdx.x * 128;
    const int m0 = blockIdx.y * 64;

    const int bn_local = tid & 127;
    const int kc = (tid >> 7) * 16;
    const int ngf = n0 + bn_local;
    const int bb = ngf / 3025;
    const int rr = ngf - bb * 3025;
    const int f_oh = rr / 55;
    const int f_ow = rr - f_oh * 55;
    const float* xb = x + (long)bb * 3 * 224 * 224;

    const int am_local = tid >> 2;
    const int akf = (tid & 3) * 8;
    const float* wrow = w1r + (m0 + am_local) * 384;

    v8f acc[2][2] = {};

    for (int k0 = 0; k0 < 384; k0 += 32) {
        {
            const float4* src = (const float4*)(wrow + k0 + akf);
            float4 v0 = src[0], v1 = src[1];
            float* d = Asf + am_local * 36 + akf;
            d[0] = v0.x; d[1] = v0.y; d[2] = v0.z; d[3] = v0.w;
            d[4] = v1.x; d[5] = v1.y; d[6] = v1.z; d[7] = v1.w;
        }
        {
            float* d = Bsf + bn_local * 36 + kc;
            #pragma unroll
            for (int i = 0; i < 16; ++i) {
                int k = k0 + kc + i;
                float v = 0.0f;
                if (k < 363) {
                    int ci = k / 121;
                    int r2 = k - ci * 121;
                    int kh = r2 / 11;
                    int kw = r2 - kh * 11;
                    int ih = f_oh * 4 - 2 + kh;
                    int iw = f_ow * 4 - 2 + kw;
                    if (ih >= 0 && ih < 224 && iw >= 0 && iw < 224)
                        v = xb[(ci * 224 + ih) * 224 + iw];
                }
                d[i] = v;
            }
        }
        __syncthreads();

        const int hs = lane >> 4;
        const float* ar0 = Asf + (wave_m * 32 + (lane & 15)) * 36;
        const float* ar1 = ar0 + 16 * 36;
        const float* br0 = Bsf + (wave_n * 32 + (lane & 15)) * 36;
        const float* br1 = br0 + 16 * 36;
        #pragma unroll
        for (int ks = 0; ks < 8; ++ks) {
            int off = ks * 4 + hs * 2;
            v2f a0 = *(const v2f*)(ar0 + off);
            v2f a1 = *(const v2f*)(ar1 + off);
            v2f b0 = *(const v2f*)(br0 + off);
            v2f b1 = *(const v2f*)(br1 + off);
            acc[0][0] = __builtin_amdgcn_wmma_f32_16x16x4_f32(false, a0, false, b0, (short)0, acc[0][0], false, false);
            acc[0][1] = __builtin_amdgcn_wmma_f32_16x16x4_f32(false, a0, false, b1, (short)0, acc[0][1], false, false);
            acc[1][0] = __builtin_amdgcn_wmma_f32_16x16x4_f32(false, a1, false, b0, (short)0, acc[1][0], false, false);
            acc[1][1] = __builtin_amdgcn_wmma_f32_16x16x4_f32(false, a1, false, b1, (short)0, acc[1][1], false, false);
        }
        __syncthreads();
    }

    const int hs = lane >> 4;
    #pragma unroll
    for (int am = 0; am < 2; ++am) {
        int m_base = m0 + wave_m * 32 + am * 16 + hs * 8;
        if (m_base >= 96) continue;
        #pragma unroll
        for (int bn = 0; bn < 2; ++bn) {
            int ng = n0 + wave_n * 32 + bn * 16 + (lane & 15);
            v8f a = acc[am][bn];
            float* op = out + (long)ng * 96 + m_base;
            float4 lo, hi;
            lo.x = a[0]; lo.y = a[1]; lo.z = a[2]; lo.w = a[3];
            hi.x = a[4]; hi.y = a[5]; hi.z = a[6]; hi.w = a[7];
            *(float4*)op = lo;
            *(float4*)(op + 4) = hi;
        }
    }
}

__global__ __launch_bounds__(256) void k_lin3_gemm(
    const float* __restrict__ X, const float* __restrict__ w3r, float* __restrict__ P) {
    __shared__ __align__(16) float Asf[64 * 36];
    __shared__ __align__(16) float Bsf[128 * 36];

    const int tid = threadIdx.x;
    const int lane = tid & 31;
    const int wid = tid >> 5;
    const int wave_m = wid & 1;
    const int wave_n = wid >> 1;
    const int seg0 = blockIdx.x * 512;
    const int m0 = blockIdx.y * 64;

    const int am_local = tid >> 2;
    const int akf = (tid & 3) * 8;
    const float* wrow = w3r + (long)(m0 + am_local) * 4096 + seg0;

    const int rb = tid & 127;
    const int kcb = (tid >> 7) * 16;
    const float* xrow = X + (long)rb * 4096 + seg0;

    v8f acc[2][2] = {};

    for (int k0 = 0; k0 < 512; k0 += 32) {
        {
            const float4* src = (const float4*)(wrow + k0 + akf);
            float4 v0 = src[0], v1 = src[1];
            float* d = Asf + am_local * 36 + akf;
            d[0] = v0.x; d[1] = v0.y; d[2] = v0.z; d[3] = v0.w;
            d[4] = v1.x; d[5] = v1.y; d[6] = v1.z; d[7] = v1.w;
        }
        {
            const float4* src = (const float4*)(xrow + k0 + kcb);
            float* d = Bsf + rb * 36 + kcb;
            #pragma unroll
            for (int j = 0; j < 4; ++j) {
                float4 v = src[j];
                d[4 * j + 0] = v.x; d[4 * j + 1] = v.y;
                d[4 * j + 2] = v.z; d[4 * j + 3] = v.w;
            }
        }
        __syncthreads();

        const int hs = lane >> 4;
        const float* ar0 = Asf + (wave_m * 32 + (lane & 15)) * 36;
        const float* ar1 = ar0 + 16 * 36;
        const float* br0 = Bsf + (wave_n * 32 + (lane & 15)) * 36;
        const float* br1 = br0 + 16 * 36;
        #pragma unroll
        for (int ks = 0; ks < 8; ++ks) {
            int off = ks * 4 + hs * 2;
            v2f a0 = *(const v2f*)(ar0 + off);
            v2f a1 = *(const v2f*)(ar1 + off);
            v2f b0 = *(const v2f*)(br0 + off);
            v2f b1 = *(const v2f*)(br1 + off);
            acc[0][0] = __builtin_amdgcn_wmma_f32_16x16x4_f32(false, a0, false, b0, (short)0, acc[0][0], false, false);
            acc[0][1] = __builtin_amdgcn_wmma_f32_16x16x4_f32(false, a0, false, b1, (short)0, acc[0][1], false, false);
            acc[1][0] = __builtin_amdgcn_wmma_f32_16x16x4_f32(false, a1, false, b0, (short)0, acc[1][0], false, false);
            acc[1][1] = __builtin_amdgcn_wmma_f32_16x16x4_f32(false, a1, false, b1, (short)0, acc[1][1], false, false);
        }
        __syncthreads();
    }

    const int hs = lane >> 4;
    float* Pseg = P + (long)blockIdx.x * 128 * 1024;
    #pragma unroll
    for (int am = 0; am < 2; ++am) {
        int m_base = m0 + wave_m * 32 + am * 16 + hs * 8;
        #pragma unroll
        for (int bn = 0; bn < 2; ++bn) {
            int ng = wave_n * 32 + bn * 16 + (lane & 15);
            v8f a = acc[am][bn];
            float* op = Pseg + (long)ng * 1024 + m_base;
            float4 lo, hi;
            lo.x = a[0]; lo.y = a[1]; lo.z = a[2]; lo.w = a[3];
            hi.x = a[4]; hi.y = a[5]; hi.z = a[6]; hi.w = a[7];
            *(float4*)op = lo;
            *(float4*)(op + 4) = hi;
        }
    }
}

__global__ void k_lin3_reduce(const float* __restrict__ P, const float* __restrict__ bias,
                              float* __restrict__ out, int total) {
    int i = blockIdx.x * 256 + threadIdx.x;
    if (i >= total) return;
    int m = i % 1000;
    int n = i / 1000;
    float acc = bias[m];
    #pragma unroll
    for (int s = 0; s < 8; ++s)
        acc += P[((long)s * 128 + n) * 1024 + m];
    out[i] = acc;
}

__global__ void k_post(const float* __restrict__ in, const float* __restrict__ bnp,
                       signed char* __restrict__ out,
                       int C, int H, int W, int HO, int WO, int pad, int dopool, int total) {
    int i = blockIdx.x * 256 + threadIdx.x;
    if (i >= total) return;
    int PW = WO + 2 * pad, PH = HO + 2 * pad;
    int c = i % C;  int t = i / C;
    int pw = t % PW; t /= PW;
    int ph = t % PH; int b = t / PH;
    signed char r;
    if (ph < pad || ph >= HO + pad || pw < pad || pw >= WO + pad) {
        r = -1;
    } else {
        int oh = ph - pad, ow = pw - pad;
        float v;
        if (dopool) {
            const float* p = in + ((long)(b * H + oh * 2) * W + ow * 2) * C + c;
            v = -INFINITY;
            #pragma unroll
            for (int dy = 0; dy < 3; ++dy)
                #pragma unroll
                for (int dx = 0; dx < 3; ++dx)
                    v = fmaxf(v, p[(dy * W + dx) * C]);
        } else {
            v = in[((long)(b * H + oh) * W + ow) * C + c];
        }
        v = bn_apply(v, bnp, C, c);
        r = (v >= 0.0f) ? (signed char)1 : (signed char)-1;
    }
    out[i] = r;
}

__device__ __forceinline__ v8i frag_a(const signed char* __restrict__ As, int row, int hs) {
    const unsigned long long* p = (const unsigned long long*)(As + row * 72);
    v8i a;
    #pragma unroll
    for (int i = 0; i < 4; ++i) {
        unsigned long long c = p[2 * hs + (i & 1) + 4 * (i >> 1)];
        a[2 * i]     = (int)(unsigned int)c;
        a[2 * i + 1] = (int)(unsigned int)(c >> 32);
    }
    return a;
}

__device__ __forceinline__ v8i frag_b(const signed char* __restrict__ Bs, int row, int hs) {
    const unsigned long long* p = (const unsigned long long*)(Bs + row * 72);
    v8i b;
    unsigned long long c;
    c = p[2 * hs];     b[0] = (int)(unsigned int)c; b[1] = (int)(unsigned int)(c >> 32);
    c = p[2 * hs + 1]; b[2] = (int)(unsigned int)c; b[3] = (int)(unsigned int)(c >> 32);
    c = p[2 * hs + 4]; b[4] = (int)(unsigned int)c; b[5] = (int)(unsigned int)(c >> 32);
    c = p[2 * hs + 5]; b[6] = (int)(unsigned int)c; b[7] = (int)(unsigned int)(c >> 32);
    return b;
}

template <int KH, int KW, int CIN>
__global__ __launch_bounds__(256) void k_bconv_gemm(
    const signed char* __restrict__ act, const signed char* __restrict__ w8,
    float* __restrict__ out,
    int PH, int PW, int Cout, int OH, int OW) {
    constexpr int K = KH * KW * CIN;
    constexpr int KWC = KW * CIN;
    __shared__ __align__(16) signed char As[64 * 72];
    __shared__ __align__(16) signed char Bs[128 * 72];

    const int tid = threadIdx.x;
    const int lane = tid & 31;
    const int wid = tid >> 5;
    const int wave_m = wid & 1;
    const int wave_n = wid >> 1;
    const int n0 = blockIdx.x * 128;
    const int m0 = blockIdx.y * 64;
    const int OHOW = OH * OW;

    const int bn_local = tid & 127;
    const int bk_base = (tid >> 7) * 32;
    const int ngf = n0 + bn_local;
    const int bb = ngf / OHOW;
    const int ppos = ngf - bb * OHOW;
    const int f_oh = ppos / OW;
    const int f_ow = ppos - f_oh * OW;
    const signed char* actb = act + (long)bb * PH * PW * CIN;

    const int am_local = tid >> 2;
    const int ak_local = (tid & 3) * 16;
    const signed char* wrow = w8 + (long)(m0 + am_local) * K;

    v8i acc[2][2] = {};

    for (int k0 = 0; k0 < K; k0 += 64) {
        {
            int kg = k0 + ak_local;
            unsigned int* dst = (unsigned int*)(As + am_local * 72 + ak_local);
            if (kg + 16 <= K) {
                const int4 v = *(const int4*)(wrow + kg);
                dst[0] = v.x; dst[1] = v.y; dst[2] = v.z; dst[3] = v.w;
            } else {
                dst[0] = 0; dst[1] = 0; dst[2] = 0; dst[3] = 0;
            }
        }
        #pragma unroll
        for (int half = 0; half < 2; ++half) {
            int kb = bk_base + half * 16;
            int kg = k0 + kb;
            unsigned int* dst = (unsigned int*)(Bs + bn_local * 72 + kb);
            if (kg + 16 <= K) {
                int kh = kg / KWC;
                int rr = kg - kh * KWC;
                int kw = rr / CIN;
                int ci = rr - kw * CIN;
                const int4 v = *(const int4*)(actb + ((f_oh + kh) * PW + (f_ow + kw)) * CIN + ci);
                dst[0] = v.x; dst[1] = v.y; dst[2] = v.z; dst[3] = v.w;
            } else {
                dst[0] = 0; dst[1] = 0; dst[2] = 0; dst[3] = 0;
            }
        }
        __syncthreads();

        const int hs = lane >> 4;
        v8i a0 = frag_a(As, wave_m * 32 + (lane & 15), hs);
        v8i a1 = frag_a(As, wave_m * 32 + 16 + (lane & 15), hs);
        v8i b0 = frag_b(Bs, wave_n * 32 + (lane & 15), hs);
        v8i b1 = frag_b(Bs, wave_n * 32 + 16 + (lane & 15), hs);
        acc[0][0] = __builtin_amdgcn_wmma_i32_16x16x64_iu8(true, a0, true, b0, acc[0][0], false, false);
        acc[0][1] = __builtin_amdgcn_wmma_i32_16x16x64_iu8(true, a0, true, b1, acc[0][1], false, false);
        acc[1][0] = __builtin_amdgcn_wmma_i32_16x16x64_iu8(true, a1, true, b0, acc[1][0], false, false);
        acc[1][1] = __builtin_amdgcn_wmma_i32_16x16x64_iu8(true, a1, true, b1, acc[1][1], false, false);
        __syncthreads();
    }

    const int hs = lane >> 4;
    #pragma unroll
    for (int am = 0; am < 2; ++am) {
        int m_base = m0 + wave_m * 32 + am * 16 + hs * 8;
        #pragma unroll
        for (int bn = 0; bn < 2; ++bn) {
            int ng = n0 + wave_n * 32 + bn * 16 + (lane & 15);
            v8i a = acc[am][bn];
            float* op = out + (long)ng * Cout + m_base;
            float4 lo, hi;
            lo.x = (float)a[0]; lo.y = (float)a[1]; lo.z = (float)a[2]; lo.w = (float)a[3];
            hi.x = (float)a[4]; hi.y = (float)a[5]; hi.z = (float)a[6]; hi.w = (float)a[7];
            *(float4*)op = lo;
            *(float4*)(op + 4) = hi;
        }
    }
}

__global__ __launch_bounds__(256) void k_blin_gemm(
    const signed char* __restrict__ X, const signed char* __restrict__ w8,
    float* __restrict__ out, int M, int K) {
    __shared__ __align__(16) signed char As[64 * 72];
    __shared__ __align__(16) signed char Bs[128 * 72];

    const int tid = threadIdx.x;
    const int lane = tid & 31;
    const int wid = tid >> 5;
    const int wave_m = wid & 1;
    const int wave_n = wid >> 1;
    const int n0 = blockIdx.x * 128;
    const int m0 = blockIdx.y * 64;

    const int rl = tid >> 2;
    const int kl = (tid & 3) * 16;
    const signed char* wrow = w8 + (long)(m0 + rl) * K;
    const int rb = tid & 127;
    const int kb = (tid >> 7) * 32;
    const signed char* xrow = X + (long)(n0 + rb) * K;

    v8i acc[2][2] = {};

    for (int k0 = 0; k0 < K; k0 += 64) {
        {
            unsigned int* da = (unsigned int*)(As + rl * 72 + kl);
            const int4 va = *(const int4*)(wrow + k0 + kl);
            da[0] = va.x; da[1] = va.y; da[2] = va.z; da[3] = va.w;
        }
        #pragma unroll
        for (int half = 0; half < 2; ++half) {
            unsigned int* db = (unsigned int*)(Bs + rb * 72 + kb + half * 16);
            const int4 vb = *(const int4*)(xrow + k0 + kb + half * 16);
            db[0] = vb.x; db[1] = vb.y; db[2] = vb.z; db[3] = vb.w;
        }
        __syncthreads();

        const int hs = lane >> 4;
        v8i a0 = frag_a(As, wave_m * 32 + (lane & 15), hs);
        v8i a1 = frag_a(As, wave_m * 32 + 16 + (lane & 15), hs);
        v8i b0 = frag_b(Bs, wave_n * 32 + (lane & 15), hs);
        v8i b1 = frag_b(Bs, wave_n * 32 + 16 + (lane & 15), hs);
        acc[0][0] = __builtin_amdgcn_wmma_i32_16x16x64_iu8(true, a0, true, b0, acc[0][0], false, false);
        acc[0][1] = __builtin_amdgcn_wmma_i32_16x16x64_iu8(true, a0, true, b1, acc[0][1], false, false);
        acc[1][0] = __builtin_amdgcn_wmma_i32_16x16x64_iu8(true, a1, true, b0, acc[1][0], false, false);
        acc[1][1] = __builtin_amdgcn_wmma_i32_16x16x64_iu8(true, a1, true, b1, acc[1][1], false, false);
        __syncthreads();
    }

    const int hs = lane >> 4;
    #pragma unroll
    for (int am = 0; am < 2; ++am) {
        int m_base = m0 + wave_m * 32 + am * 16 + hs * 8;
        #pragma unroll
        for (int bn = 0; bn < 2; ++bn) {
            int ng = n0 + wave_n * 32 + bn * 16 + (lane & 15);
            v8i a = acc[am][bn];
            float* op = out + (long)ng * M + m_base;
            float4 lo, hi;
            lo.x = (float)a[0]; lo.y = (float)a[1]; lo.z = (float)a[2]; lo.w = (float)a[3];
            hi.x = (float)a[4]; hi.y = (float)a[5]; hi.z = (float)a[6]; hi.w = (float)a[7];
            *(float4*)op = lo;
            *(float4*)(op + 4) = hi;
        }
    }
}

__global__ void k_bnbin_feat(const float* __restrict__ in, const float* __restrict__ bnp,
                             signed char* __restrict__ out, int M, int total) {
    int i = blockIdx.x * 256 + threadIdx.x;
    if (i >= total) return;
    int c = i % M;
    float y = bn_apply(in[i], bnp, M, c);
    out[i] = (y >= 0.0f) ? (signed char)1 : (signed char)-1;
}

__global__ void k_bnrelu_feat(const float* __restrict__ in, const float* __restrict__ bnp,
                              float* __restrict__ out, int M, int total) {
    int i = blockIdx.x * 256 + threadIdx.x;
    if (i >= total) return;
    int c = i % M;
    float y = bn_apply(in[i], bnp, M, c);
    out[i] = fmaxf(y, 0.0f);
}

extern "C" void kernel_launch(void* const* d_in, const int* in_sizes, int n_in,
                              void* d_out, int out_size, void* d_ws, size_t ws_size,
                              hipStream_t stream) {
    const float* x        = (const float*)d_in[0];
    const float* conv1_w  = (const float*)d_in[1];
    const float* bconv2_w = (const float*)d_in[2];
    const float* bconv3_w = (const float*)d_in[3];
    const float* bconv4_w = (const float*)d_in[4];
    const float* bconv5_w = (const float*)d_in[5];
    const float* blin1_w  = (const float*)d_in[6];
    const float* blin2_w  = (const float*)d_in[7];
    const float* lin3_w   = (const float*)d_in[8];
    const float* lin3_b   = (const float*)d_in[9];
    const float* bn1 = (const float*)d_in[10];
    const float* bn2 = (const float*)d_in[11];
    const float* bn3 = (const float*)d_in[12];
    const float* bn4 = (const float*)d_in[13];
    const float* bn5 = (const float*)d_in[14];
    const float* bn6 = (const float*)d_in[15];
    const float* bn7 = (const float*)d_in[16];

    char* ws = (char*)d_ws;
    size_t off = 0;
    auto alloc = [&](size_t bytes) -> void* {
        void* p = ws + off;
        off = (off + bytes + 255) & ~(size_t)255;
        return p;
    };

    float* F = (float*)alloc((size_t)128 * 55 * 55 * 96 * sizeof(float));
    signed char* A0 = (signed char*)alloc(12u * 1024 * 1024);
    signed char* A1 = (signed char*)alloc(12u * 1024 * 1024);
    float* w1r = (float*)alloc((size_t)128 * 384 * sizeof(float));
    float* w3r = (float*)alloc((size_t)1024 * 4096 * sizeof(float));
    float* P3 = (float*)alloc((size_t)8 * 128 * 1024 * sizeof(float));
    signed char* w2_8  = (signed char*)alloc((size_t)256 * 2400);
    signed char* w3_8  = (signed char*)alloc((size_t)384 * 2304);
    signed char* w4_8  = (signed char*)alloc((size_t)384 * 3456);
    signed char* w5_8  = (signed char*)alloc((size_t)256 * 3456);
    signed char* wl1_8 = (signed char*)alloc((size_t)4096 * 9216);
    signed char* wl2_8 = (signed char*)alloc((size_t)4096 * 4096);
    float*       L1 = (float*)alloc((size_t)128 * 4096 * sizeof(float));
    signed char* A6 = (signed char*)alloc((size_t)128 * 4096);
    float*       L2 = (float*)alloc((size_t)128 * 4096 * sizeof(float));
    float*       R  = (float*)alloc((size_t)128 * 4096 * sizeof(float));

    auto up = [](int n) { return (n + 255) / 256; };

    k_prep_w1<<<up(128 * 384), 256, 0, stream>>>(conv1_w, w1r, 128 * 384);
    k_prep_w3<<<up(1024 * 4096), 256, 0, stream>>>(lin3_w, w3r, 1024 * 4096);
    k_binw_conv<<<up(256 * 2400), 256, 0, stream>>>(bconv2_w, w2_8, 96, 5, 5, 256 * 2400);
    k_binw_conv<<<up(384 * 2304), 256, 0, stream>>>(bconv3_w, w3_8, 256, 3, 3, 384 * 2304);
    k_binw_conv<<<up(384 * 3456), 256, 0, stream>>>(bconv4_w, w4_8, 384, 3, 3, 384 * 3456);
    k_binw_conv<<<up(256 * 3456), 256, 0, stream>>>(bconv5_w, w5_8, 384, 3, 3, 256 * 3456);
    k_binw_lin1<<<up(4096 * 9216), 256, 0, stream>>>(blin1_w, wl1_8, 4096 * 9216);
    k_binw<<<up(4096 * 4096), 256, 0, stream>>>(blin2_w, wl2_8, 4096 * 4096);

    k_conv1_gemm<<<dim3(387200 / 128, 128 / 64), 256, 0, stream>>>(x, w1r, F);

    int p1_total = 128 * 31 * 31 * 96;
    k_post<<<up(p1_total), 256, 0, stream>>>(F, bn1, A0, 96, 55, 55, 27, 27, 2, 1, p1_total);

    k_bconv_gemm<5, 5, 96><<<dim3(93312 / 128, 256 / 64), 256, 0, stream>>>(
        A0, w2_8, F, 31, 31, 256, 27, 27);

    int p2_total = 128 * 15 * 15 * 256;
    k_post<<<up(p2_total), 256, 0, stream>>>(F, bn2, A1, 256, 27, 27, 13, 13, 1, 1, p2_total);

    k_bconv_gemm<3, 3, 256><<<dim3(21632 / 128, 384 / 64), 256, 0, stream>>>(
        A1, w3_8, F, 15, 15, 384, 13, 13);

    int p3_total = 128 * 15 * 15 * 384;
    k_post<<<up(p3_total), 256, 0, stream>>>(F, bn3, A0, 384, 13, 13, 13, 13, 1, 0, p3_total);

    k_bconv_gemm<3, 3, 384><<<dim3(21632 / 128, 384 / 64), 256, 0, stream>>>(
        A0, w4_8, F, 15, 15, 384, 13, 13);

    k_post<<<up(p3_total), 256, 0, stream>>>(F, bn4, A1, 384, 13, 13, 13, 13, 1, 0, p3_total);

    k_bconv_gemm<3, 3, 384><<<dim3(21632 / 128, 256 / 64), 256, 0, stream>>>(
        A1, w5_8, F, 15, 15, 256, 13, 13);

    int p5_total = 128 * 6 * 6 * 256;
    k_post<<<up(p5_total), 256, 0, stream>>>(F, bn5, A0, 256, 13, 13, 6, 6, 0, 1, p5_total);

    k_blin_gemm<<<dim3(1, 4096 / 64), 256, 0, stream>>>(A0, wl1_8, L1, 4096, 9216);

    k_bnbin_feat<<<up(128 * 4096), 256, 0, stream>>>(L1, bn6, A6, 4096, 128 * 4096);

    k_blin_gemm<<<dim3(1, 4096 / 64), 256, 0, stream>>>(A6, wl2_8, L2, 4096, 4096);

    k_bnrelu_feat<<<up(128 * 4096), 256, 0, stream>>>(L2, bn7, R, 4096, 128 * 4096);

    k_lin3_gemm<<<dim3(8, 16), 256, 0, stream>>>(R, w3r, P3);
    k_lin3_reduce<<<up(128 * 1000), 256, 0, stream>>>(P3, lin3_b, (float*)d_out, 128 * 1000);
}
